// ODEVRNN_65695819760243
// MI455X (gfx1250) — matrix operand order repaired and checked by reading, not compiled, not run on hardware
//
#include <hip/hip_runtime.h>

typedef __attribute__((ext_vector_type(16))) _Float16 v16h;
typedef __attribute__((ext_vector_type(8)))  _Float16 v8h;
typedef __attribute__((ext_vector_type(8)))  float    v8f;

#define LSEQ 256
#define BSZ  1024

// ---------------- weight packing: f32 W(out,in) -> f16 WMMA B-fragments -------------
// Fragment (kt,nt) = 32 lanes x 16 halves, lane l covers column n = nt*16+(l&15),
// K halves k = kt*32 + (i>>3)*16 + (l>>4)*8 + (i&7)  (16-bit B-matrix layout, 32x16 per fragment).
__global__ void pack_w_kernel(const float* __restrict__ W, _Float16* __restrict__ dst,
                              int IN, int OUT) {
  int total = IN * OUT;
  int tid = blockIdx.x * blockDim.x + threadIdx.x;
  if (tid >= total) return;
  int KT   = IN >> 5;
  int frag = tid >> 9;          // /512
  int rem  = tid & 511;
  int lane = rem >> 4;
  int i    = rem & 15;
  int kt   = frag % KT;
  int nt   = frag / KT;
  int n = nt * 16 + (lane & 15);
  int k = kt * 32 + ((i >> 3) * 16) + ((lane >> 4) * 8) + (i & 7);
  dst[tid] = (_Float16)W[n * IN + k];
}

// ---------------- device helpers -------------------------------------------------
__device__ __forceinline__ float sigm_(float x) { return 1.f / (1.f + __expf(-x)); }

// A-fragment (16x32 f16) from a row-major [16][strideH] f16 LDS buffer.
// Lane l: row m=l&15, halves 0..7 -> k = kt*32 + (l>>4)*8 + i, halves 8..15 -> +16.
__device__ __forceinline__ v16h load_afrag(const _Float16* base, int strideH, int kt) {
  const int l = threadIdx.x & 31;
  const _Float16* p = base + (l & 15) * strideH + kt * 32 + ((l >> 4) * 8);
  v8h lo = *(const v8h*)p;
  v8h hi = *(const v8h*)(p + 16);
  v16h r;
#pragma unroll
  for (int i = 0; i < 8; ++i) { r[i] = lo[i]; r[i + 8] = hi[i]; }
  return r;
}

// D = act(A[16xIN] * W + b); epilogue gets C accumulators per 16-col tile.
template <int IN, int OUT, typename EPI>
__device__ __forceinline__ void gemm(const _Float16* aLDS, int aStrideH,
                                     const _Float16* wfrag, const float* bias, EPI epi) {
  const int l = threadIdx.x & 31;
  constexpr int KT = IN / 32;
  constexpr int NT = OUT / 16;
  for (int nt = 0; nt < NT; ++nt) {
    float bv = bias[nt * 16 + (l & 15)];
    v8f acc;
#pragma unroll
    for (int v = 0; v < 8; ++v) acc[v] = bv;
#pragma unroll
    for (int kt = 0; kt < KT; ++kt) {
      v16h a = load_afrag(aLDS, aStrideH, kt);
      v16h b = ((const v16h*)wfrag)[(nt * KT + kt) * 32 + l];
      acc = __builtin_amdgcn_wmma_f32_16x16x32_f16(false, a, false, b, (short)0, acc,
                                                   false, false);
    }
    epi(nt, acc);
  }
}

// ws half-offsets of packed weights (dict order; ode1/ode2 contiguous at the end)
#define OFF_PU1    0
#define OFF_PU2    2048
#define OFF_PX1    6144
#define OFF_PX2    8192
#define OFF_PZ1    12288
#define OFF_PZ2    16384
#define OFF_POST1  20480
#define OFF_POST2  57344
#define OFF_PMU    94208
#define OFF_PLS    106496
#define OFF_GIH    118784
#define OFF_GHH    155648
#define OFF_ODE1   167936
#define OFF_ODE2   176128

#define ODE_LDS_BYTES 32768
#define WAVE_BYTES    53312
#define SMEM_BYTES    (ODE_LDS_BYTES + 2 * WAVE_BYTES)

__global__ void __launch_bounds__(64)
odevrnn_main(const float* __restrict__ ext, const float* __restrict__ obsIn,
             const float* __restrict__ epsIn, const _Float16* __restrict__ wp,
             const float* b_pu1, const float* b_pu2, const float* b_px1, const float* b_px2,
             const float* b_pz1, const float* b_pz2, const float* b_post1, const float* b_post2,
             const float* b_pmu, const float* b_pls, const float* b_gih, const float* b_ghh,
             const float* b_o1, const float* b_o2, float* __restrict__ out) {
  extern __shared__ char smem_raw[];
  const int tid = threadIdx.x;
  const int l = tid & 31;
  const int w = tid >> 5;
  const int b0 = (blockIdx.x * 2 + w) * 16;

  // Shared (block-wide) LDS copy of ODE weight fragments, staged asynchronously.
  const _Float16* odeW1 = (const _Float16*)smem_raw;
  const _Float16* odeW2 = (const _Float16*)(smem_raw + 16384);
  {
    unsigned long long src = (unsigned long long)(wp + OFF_ODE1);
    unsigned ldsb = (unsigned)(unsigned long long)smem_raw; // flat->LDS offset (low 32b)
#pragma unroll 1
    for (int it = 0; it < 32; ++it) {
      unsigned boff = (unsigned)((tid + it * 64) * 16);
      unsigned la = ldsb + boff;
      asm volatile("global_load_async_to_lds_b128 %0, %1, %2 offset:0"
                   :: "v"(la), "v"(boff), "s"(src) : "memory");
    }
    asm volatile("s_wait_asynccnt 0" ::: "memory");
    __syncthreads();
  }

  // Per-wave LDS scratch
  char* wb = smem_raw + ODE_LDS_BYTES + w * WAVE_BYTES;
  _Float16* catA  = (_Float16*)(wb + 0);      // 16x192 : [x_e | u_e | h -> z_e]
  _Float16* bufT  = (_Float16*)(wb + 6144);   // 16x192 : dblock gate t (also 16x64 z1)
  _Float16* bufIn = (_Float16*)(wb + 12288);  // 16x128 : raw input / ODE tanh layer
  _Float16* Hin16 = (_Float16*)(wb + 16384);  // 16x64  : embed tmp / z / ODE h input
  _Float16* bufH  = (_Float16*)(wb + 18432);  // 16x64  : persistent h (f16)
  float*    Hb    = (float*)(wb + 20480);     // 16x64  : ODE base h (f32)
  float*    AccO  = (float*)(wb + 24576);     // 16x64  : RK4 accumulator
  float*    F32A  = (float*)(wb + 28672);     // 16x192 f32 scratch
  float*    F32B  = (float*)(wb + 40960);     // 16x192 f32 scratch
  float*    dtbuf = (float*)(wb + 53248);     // 16 x f32

  // weight fragment pointers (global, L2-resident, ~360KB total)
  const _Float16* Wpu1 = wp + OFF_PU1;  const _Float16* Wpu2 = wp + OFF_PU2;
  const _Float16* Wpx1 = wp + OFF_PX1;  const _Float16* Wpx2 = wp + OFF_PX2;
  const _Float16* Wpz1 = wp + OFF_PZ1;  const _Float16* Wpz2 = wp + OFF_PZ2;
  const _Float16* Wp1  = wp + OFF_POST1; const _Float16* Wp2 = wp + OFF_POST2;
  const _Float16* Wmu  = wp + OFF_PMU;  const _Float16* Wls = wp + OFF_PLS;
  const _Float16* Wgih = wp + OFF_GIH;  const _Float16* Wghh = wp + OFF_GHH;

  // h0 = 0
#pragma unroll 1
  for (int it = 0; it < 32; ++it) bufH[l + it * 32] = (_Float16)0.f;

#pragma unroll 1
  for (int tstep = 0; tstep < LSEQ; ++tstep) {
    const int m16 = l & 15, cb = l >> 4;
    const size_t rowBase = (size_t)tstep * BSZ + b0;

    // ---- x_e = preprocess(obs) ----
    {
      const float* row = obsIn + (rowBase + m16) * 32 + cb * 16;
#pragma unroll
      for (int i = 0; i < 16; ++i) bufIn[m16 * 32 + cb * 16 + i] = (_Float16)row[i];
    }
    gemm<32, 64>(bufIn, 32, Wpx1, b_px1, [&](int nt, v8f acc) {
#pragma unroll
      for (int v = 0; v < 8; ++v) {
        int m = v + 8 * (l >> 4), n = nt * 16 + (l & 15);
        Hin16[m * 64 + n] = (_Float16)fmaxf(acc[v], 0.f);
      }
    });
    gemm<64, 64>(Hin16, 64, Wpx2, b_px2, [&](int nt, v8f acc) {
#pragma unroll
      for (int v = 0; v < 8; ++v) {
        int m = v + 8 * (l >> 4), n = nt * 16 + (l & 15);
        catA[m * 192 + n] = (_Float16)fmaxf(acc[v], 0.f);
      }
    });

    // ---- u_e = preprocess(u), dt ----
    {
      const float* row = ext + (rowBase + m16) * 33 + cb * 16;
#pragma unroll
      for (int i = 0; i < 16; ++i) bufIn[m16 * 32 + cb * 16 + i] = (_Float16)row[i];
      if (l < 16) dtbuf[l] = ext[(rowBase + l) * 33 + 32];
    }
    gemm<32, 64>(bufIn, 32, Wpu1, b_pu1, [&](int nt, v8f acc) {
#pragma unroll
      for (int v = 0; v < 8; ++v) {
        int m = v + 8 * (l >> 4), n = nt * 16 + (l & 15);
        Hin16[m * 64 + n] = (_Float16)fmaxf(acc[v], 0.f);
      }
    });
    gemm<64, 64>(Hin16, 64, Wpu2, b_pu2, [&](int nt, v8f acc) {
#pragma unroll
      for (int v = 0; v < 8; ++v) {
        int m = v + 8 * (l >> 4), n = nt * 16 + (l & 15);
        catA[m * 192 + 64 + n] = (_Float16)fmaxf(acc[v], 0.f);
      }
    });

    // ---- cat[...,128:192] = h ----
#pragma unroll 1
    for (int it = 0; it < 32; ++it) {
      int idx = l + it * 32;
      catA[(idx >> 6) * 192 + 128 + (idx & 63)] = bufH[idx];
    }

    // ---- dblock: t = tanh(post1(cat)) * sigmoid(post2(cat)) ----
    gemm<192, 192>(catA, 192, Wp1, b_post1, [&](int nt, v8f acc) {
#pragma unroll
      for (int v = 0; v < 8; ++v) {
        int m = v + 8 * (l >> 4), n = nt * 16 + (l & 15);
        F32A[m * 192 + n] = acc[v];
      }
    });
    gemm<192, 192>(catA, 192, Wp2, b_post2, [&](int nt, v8f acc) {
#pragma unroll
      for (int v = 0; v < 8; ++v) {
        int m = v + 8 * (l >> 4), n = nt * 16 + (l & 15);
        bufT[m * 192 + n] = (_Float16)(tanhf(F32A[m * 192 + n]) * sigm_(acc[v]));
      }
    });
    // mu, logsig, z
    gemm<192, 64>(bufT, 192, Wmu, b_pmu, [&](int nt, v8f acc) {
#pragma unroll
      for (int v = 0; v < 8; ++v) {
        int m = v + 8 * (l >> 4), n = nt * 16 + (l & 15);
        F32A[m * 64 + n] = acc[v];
        out[(rowBase + m) * 256 + n] = acc[v];
      }
    });
    gemm<192, 64>(bufT, 192, Wls, b_pls, [&](int nt, v8f acc) {
#pragma unroll
      for (int v = 0; v < 8; ++v) {
        int m = v + 8 * (l >> 4), n = nt * 16 + (l & 15);
        size_t r = rowBase + m;
        float ls = acc[v];
        out[r * 256 + 64 + n] = ls;
        float z = F32A[m * 64 + n] + __expf(ls) * epsIn[r * 64 + n];
        out[r * 256 + 128 + n] = z;
        Hin16[m * 64 + n] = (_Float16)z; // z as next GEMM input
      }
    });

    // ---- z_e = preprocess(z) -> cat[...,128:192] ----
    gemm<64, 64>(Hin16, 64, Wpz1, b_pz1, [&](int nt, v8f acc) {
#pragma unroll
      for (int v = 0; v < 8; ++v) {
        int m = v + 8 * (l >> 4), n = nt * 16 + (l & 15);
        bufT[m * 64 + n] = (_Float16)fmaxf(acc[v], 0.f);
      }
    });
    gemm<64, 64>(bufT, 64, Wpz2, b_pz2, [&](int nt, v8f acc) {
#pragma unroll
      for (int v = 0; v < 8; ++v) {
        int m = v + 8 * (l >> 4), n = nt * 16 + (l & 15);
        catA[m * 192 + 128 + n] = (_Float16)fmaxf(acc[v], 0.f);
      }
    });

    // ---- ODE evolve: 4 RK4 steps, f(h) = ode2(tanh(ode1 h)), weights in LDS ----
#pragma unroll 1
    for (int it = 0; it < 32; ++it) {
      int idx = l + it * 32;
      float hv = (float)bufH[idx];
      Hb[idx] = hv;
      Hin16[idx] = bufH[idx];
    }
#pragma unroll 1
    for (int os = 0; os < 4; ++os) {
#pragma unroll 1
      for (int st = 0; st < 4; ++st) {
        gemm<64, 128>(Hin16, 64, odeW1, b_o1, [&](int nt, v8f acc) {
#pragma unroll
          for (int v = 0; v < 8; ++v) {
            int m = v + 8 * (l >> 4), n = nt * 16 + (l & 15);
            bufIn[m * 128 + n] = (_Float16)tanhf(acc[v]);
          }
        });
        gemm<128, 64>(bufIn, 128, odeW2, b_o2, [&](int nt, v8f acc) {
#pragma unroll
          for (int v = 0; v < 8; ++v) {
            int m = v + 8 * (l >> 4), n = nt * 16 + (l & 15);
            int idx = m * 64 + n;
            float k = acc[v];
            float sdt = dtbuf[m] * 0.25f;
            float wk = (st == 1 || st == 2) ? 2.f : 1.f;
            float a = (st == 0) ? k : (AccO[idx] + wk * k);
            AccO[idx] = a;
            if (st < 2) {
              Hin16[idx] = (_Float16)(Hb[idx] + 0.5f * sdt * k);
            } else if (st == 2) {
              Hin16[idx] = (_Float16)(Hb[idx] + sdt * k);
            } else {
              float hn = Hb[idx] + (sdt / 6.f) * a;
              Hb[idx] = hn;
              Hin16[idx] = (_Float16)hn;
            }
          }
        });
      }
    }

    // ---- GRU: gi = gru_ih(cat[x_e,u_e,z_e]), gh = gru_hh(h_ode) ----
    gemm<192, 192>(catA, 192, Wgih, b_gih, [&](int nt, v8f acc) {
#pragma unroll
      for (int v = 0; v < 8; ++v) {
        int m = v + 8 * (l >> 4), n = nt * 16 + (l & 15);
        F32A[m * 192 + n] = acc[v];
      }
    });
    gemm<64, 192>(Hin16, 64, Wghh, b_ghh, [&](int nt, v8f acc) {
#pragma unroll
      for (int v = 0; v < 8; ++v) {
        int m = v + 8 * (l >> 4), n = nt * 16 + (l & 15);
        F32B[m * 192 + n] = acc[v];
      }
    });
#pragma unroll 1
    for (int it = 0; it < 32; ++it) {
      int idx = l + it * 32;
      int m = idx >> 6, c = idx & 63;
      float ir = F32A[m * 192 + c], iz = F32A[m * 192 + 64 + c], in = F32A[m * 192 + 128 + c];
      float hr = F32B[m * 192 + c], hz = F32B[m * 192 + 64 + c], hn = F32B[m * 192 + 128 + c];
      float r = sigm_(ir + hr);
      float zt = sigm_(iz + hz);
      float nn = tanhf(in + r * hn);
      float hnew = (1.f - zt) * nn + zt * Hb[idx];
      bufH[idx] = (_Float16)hnew;
      out[(rowBase + m) * 256 + 192 + c] = hnew;
    }
  }
}

// ---------------- host launcher ---------------------------------------------------
extern "C" void kernel_launch(void* const* d_in, const int* in_sizes, int n_in,
                              void* d_out, int out_size, void* d_ws, size_t ws_size,
                              hipStream_t stream) {
  (void)in_sizes; (void)n_in; (void)out_size; (void)ws_size;
  const float* ext = (const float*)d_in[0];
  const float* obs = (const float*)d_in[1];
  const float* eps = (const float*)d_in[2];
  _Float16* wsH = (_Float16*)d_ws;

  struct WDesc { int idx; int in; int out; size_t off; };
  static const WDesc WD[14] = {
      {3, 32, 64, OFF_PU1},    {5, 64, 64, OFF_PU2},
      {7, 32, 64, OFF_PX1},    {9, 64, 64, OFF_PX2},
      {11, 64, 64, OFF_PZ1},   {13, 64, 64, OFF_PZ2},
      {15, 192, 192, OFF_POST1}, {17, 192, 192, OFF_POST2},
      {19, 192, 64, OFF_PMU},  {21, 192, 64, OFF_PLS},
      {23, 192, 192, OFF_GIH}, {25, 64, 192, OFF_GHH},
      {27, 64, 128, OFF_ODE1}, {29, 128, 64, OFF_ODE2}};
  for (int i = 0; i < 14; ++i) {
    int total = WD[i].in * WD[i].out;
    pack_w_kernel<<<(total + 255) / 256, 256, 0, stream>>>(
        (const float*)d_in[WD[i].idx], wsH + WD[i].off, WD[i].in, WD[i].out);
  }

  odevrnn_main<<<dim3(32), dim3(64), SMEM_BYTES, stream>>>(
      ext, obs, eps, wsH,
      (const float*)d_in[4], (const float*)d_in[6], (const float*)d_in[8],
      (const float*)d_in[10], (const float*)d_in[12], (const float*)d_in[14],
      (const float*)d_in[16], (const float*)d_in[18], (const float*)d_in[20],
      (const float*)d_in[22], (const float*)d_in[24], (const float*)d_in[26],
      (const float*)d_in[28], (const float*)d_in[30], (float*)d_out);
}
